// PointRemover_40166534152223
// MI455X (gfx1250) — matrix operand order repaired and checked by reading, not compiled, not run on hardware
//
#include <hip/hip_runtime.h>
#include <hip/hip_bf16.h>

// ---------------------------------------------------------------------------
// PointRemover forward on MI455X (gfx1250, wave32, WMMA).
// All GEMM-shaped work runs on v_wmma_f32_16x16x32_f16 (f16 in, f32 accum).
// Cross-lane reductions use v_permlane16/v_permlanex16 (VALU) instead of
// ds_bpermute.
// ---------------------------------------------------------------------------

typedef _Float16 h16;
typedef __attribute__((ext_vector_type(16))) _Float16 v16h;
typedef __attribute__((ext_vector_type(8)))  float    v8f;

union F16x16 { v16h v; uint4 q[2]; };

__device__ __forceinline__ v8f wmma_f16(const v16h& a, const v16h& b, const v8f& c) {
  // 8 args: (neg_a, A, neg_b, B, c_mod, C, reuse_a, reuse_b)
  return __builtin_amdgcn_wmma_f32_16x16x32_f16(false, a, false, b, (short)0, c,
                                                false, false);
}

__device__ __forceinline__ float gelu_exact(float x) {
  return 0.5f * x * (1.0f + erff(x * 0.70710678118654752f));
}

#define ZERO8 {0.f,0.f,0.f,0.f,0.f,0.f,0.f,0.f}

// ---------------------------------------------------------------------------
// XOR-butterfly lane exchange. M in {1,2,4,8} stays inside a 16-lane group
// (v_permlane16_b32); M == 16 crosses the halves (v_permlanex16_b32).
// Selector nibbles: sel[k] = k ^ M, packed little-endian.
// ---------------------------------------------------------------------------
template <int M>
__device__ __forceinline__ float xor_f32(float v) {
  union { float f; unsigned u; } in, out;
  in.f = v;
#if __has_builtin(__builtin_amdgcn_permlane16) && __has_builtin(__builtin_amdgcn_permlanex16)
  unsigned s0, s1;
  if constexpr (M == 1)       { s0 = 0x67452301u; s1 = 0xefcdab89u; }
  else if constexpr (M == 2)  { s0 = 0x54761032u; s1 = 0xdcfe98bau; }
  else if constexpr (M == 4)  { s0 = 0x32107654u; s1 = 0xba98fedcu; }
  else if constexpr (M == 8)  { s0 = 0xfedcba98u; s1 = 0x76543210u; }
  else                        { s0 = 0x76543210u; s1 = 0xfedcba98u; } // identity
  if constexpr (M == 16)
    out.u = __builtin_amdgcn_permlanex16(in.u, in.u, s0, s1, false, false);
  else
    out.u = __builtin_amdgcn_permlane16(in.u, in.u, s0, s1, false, false);
#else
  out.f = __shfl_xor(in.f, M, 32);
#endif
  return out.f;
}

__device__ __forceinline__ float red_sum32(float v) {
  v += xor_f32<1>(v); v += xor_f32<2>(v); v += xor_f32<4>(v);
  v += xor_f32<8>(v); v += xor_f32<16>(v);
  return v;
}
__device__ __forceinline__ float red_max16(float v) {
  v = fmaxf(v, xor_f32<1>(v)); v = fmaxf(v, xor_f32<2>(v));
  v = fmaxf(v, xor_f32<4>(v)); v = fmaxf(v, xor_f32<8>(v));
  return v;
}
__device__ __forceinline__ float red_sum16(float v) {
  v += xor_f32<1>(v); v += xor_f32<2>(v);
  v += xor_f32<4>(v); v += xor_f32<8>(v);
  return v;
}

// ---------------------------------------------------------------------------
// Weight transpose + f32->f16 convert:  dst[C x R] = (h16) src[R x C]^T
// ---------------------------------------------------------------------------
__global__ void transpose_to_f16(const float* __restrict__ src,
                                 h16* __restrict__ dst, int R, int C) {
  int i = blockIdx.x * 256 + threadIdx.x;
  if (i >= R * C) return;
  int r = i / C, c = i % C;
  dst[(long)c * R + r] = (h16)src[i];
}

// ---------------------------------------------------------------------------
// Encoder embed: per row: x(3) @ W1(3x256) + b1 -> LN -> GELU -> f16.
// One wave per row; 8 waves per block.
// ---------------------------------------------------------------------------
__global__ __launch_bounds__(256)
void enc_embed_kernel(const float* __restrict__ pts,
                      const float* __restrict__ w1, const float* __restrict__ b1,
                      const float* __restrict__ lng, const float* __restrict__ lnb,
                      h16* __restrict__ y16) {
  int w = threadIdx.x >> 5, lane = threadIdx.x & 31;
  long row = (long)blockIdx.x * 8 + w;
  const float* xp = pts + row * 3;
  float x0 = xp[0], x1 = xp[1], x2 = xp[2];
  float v[8];
  float s = 0.f;
#pragma unroll
  for (int i = 0; i < 8; ++i) {
    int c = lane + 32 * i;
    v[i] = x0 * w1[c] + x1 * w1[256 + c] + x2 * w1[512 + c] + b1[c];
    s += v[i];
  }
  s = red_sum32(s);
  float mu = s * (1.0f / 256.0f);
  float qv = 0.f;
#pragma unroll
  for (int i = 0; i < 8; ++i) { float d = v[i] - mu; qv += d * d; }
  qv = red_sum32(qv);
  float rstd = rsqrtf(qv * (1.0f / 256.0f) + 1e-5f);
#pragma unroll
  for (int i = 0; i < 8; ++i) {
    int c = lane + 32 * i;
    float t = (v[i] - mu) * rstd * lng[c] + lnb[c];
    y16[row * 256 + c] = (h16)gelu_exact(t);
  }
}

// ---------------------------------------------------------------------------
// LayerNorm over 256 dims: y16 = f16(LN(h)*g + b). One wave per row.
// ---------------------------------------------------------------------------
__global__ __launch_bounds__(256)
void ln_kernel(const float* __restrict__ h,
               const float* __restrict__ g, const float* __restrict__ b,
               h16* __restrict__ y16) {
  int w = threadIdx.x >> 5, lane = threadIdx.x & 31;
  long row = (long)blockIdx.x * 8 + w;
  const float* x = h + row * 256;
  float v[8];
  float s = 0.f;
#pragma unroll
  for (int i = 0; i < 8; ++i) { v[i] = x[lane + 32 * i]; s += v[i]; }
  s = red_sum32(s);
  float mu = s * (1.0f / 256.0f);
  float qv = 0.f;
#pragma unroll
  for (int i = 0; i < 8; ++i) { float d = v[i] - mu; qv += d * d; }
  qv = red_sum32(qv);
  float rstd = rsqrtf(qv * (1.0f / 256.0f) + 1e-5f);
#pragma unroll
  for (int i = 0; i < 8; ++i) {
    int c = lane + 32 * i;
    y16[row * 256 + c] = (h16)((v[i] - mu) * rstd * g[c] + b[c]);
  }
}

// ---------------------------------------------------------------------------
// Generic batched WMMA GEMM: C[b] = A[b](M x K, f16, lda) @ Bt^T  (+ bias)
//   Bt = weight stored transposed: (N x K) row-major f16, shared over batch.
// Block = 256 threads (8 waves). Tile 128(M) x 64(N), K-step 32.
// Double-buffered LDS with register staging: one barrier per K-step, global
// loads for step i+1 overlap the WMMA work of step i.
// EPI: 0 = Cf32 = acc+bias         1 = Cf32 += acc+bias (residual)
//      2 = Ch16 = gelu(acc+bias)   3 = QKV split (q,k row-major; v transposed)
// ---------------------------------------------------------------------------
template <int EPI>
__global__ __launch_bounds__(256)
void gemm_wmma_kernel(const h16* __restrict__ A, int lda, long strideA,
                      const h16* __restrict__ Bt, int K,
                      const float* __restrict__ bias,
                      float* __restrict__ Cf, h16* __restrict__ Ch,
                      int ldc, long strideC,
                      h16* __restrict__ q16, h16* __restrict__ k16,
                      h16* __restrict__ vT) {
  __shared__ h16 sA[2][128 * 40];  // 128 rows x 32 halfs, padded stride 40
  __shared__ h16 sB[2][64 * 40];   // 64  rows x 32 halfs, padded stride 40

  const int tid = threadIdx.x;
  const int w = tid >> 5, lane = tid & 31;
  const int hf = lane >> 4, lm = lane & 15;
  const int bl = blockIdx.z;

  const h16* Ab = A + (long)bl * strideA + (long)blockIdx.x * 128 * lda;
  const h16* Bb = Bt + (long)blockIdx.y * 64 * K;

  const int srow = tid >> 2, sc4 = tid & 3;  // staging row / 16B chunk

  uint4 ar0, ar1, br;
  auto gload = [&](int kc) {
    ar0 = *(const uint4*)&Ab[(long)srow * lda + kc + sc4 * 8];
    ar1 = *(const uint4*)&Ab[(long)(srow + 64) * lda + kc + sc4 * 8];
    br  = *(const uint4*)&Bb[(long)srow * K + kc + sc4 * 8];
  };
  auto sstore = [&](int buf) {
    *(uint4*)&sA[buf][srow * 40 + sc4 * 8] = ar0;
    *(uint4*)&sA[buf][(srow + 64) * 40 + sc4 * 8] = ar1;
    *(uint4*)&sB[buf][srow * 40 + sc4 * 8] = br;
  };

  v8f zero = ZERO8;
  v8f acc[4] = {zero, zero, zero, zero};

  const int nk = K >> 5;
  gload(0);
  sstore(0);
  __syncthreads();

  for (int i = 0; i < nk; ++i) {
    if (i + 1 < nk) gload((i + 1) << 5);
    if (i + 2 < nk) {
      __builtin_prefetch(&Ab[(long)srow * lda + ((i + 2) << 5)], 0, 1);
      __builtin_prefetch(&Bb[(long)srow * K + ((i + 2) << 5)], 0, 1);
    }
    const h16* sAc = sA[i & 1];
    const h16* sBc = sB[i & 1];
    // A fragment (16x32 f16): lane lm = row; k = hf*8 + {0..7} then +16.
    F16x16 a;
    const h16* ap = &sAc[(w * 16 + lm) * 40 + hf * 8];
    a.q[0] = *(const uint4*)ap;
    a.q[1] = *(const uint4*)(ap + 16);
#pragma unroll
    for (int t = 0; t < 4; ++t) {
      // B fragment (32x16): lane lm = column n; k = hf*8 + {0..7} then +16.
      F16x16 bfr;
      const h16* bp = &sBc[(t * 16 + lm) * 40 + hf * 8];
      bfr.q[0] = *(const uint4*)bp;
      bfr.q[1] = *(const uint4*)(bp + 16);
      acc[t] = wmma_f16(a.v, bfr.v, acc[t]);
    }
    if (i + 1 < nk) sstore((i + 1) & 1);
    __syncthreads();
  }

  // Epilogue. C fragment: lane lm = n, rows m = r + 8*hf.
  const int mBase = blockIdx.x * 128 + w * 16 + hf * 8;
  const int nBase = blockIdx.y * 64 + lm;
#pragma unroll
  for (int t = 0; t < 4; ++t) {
    int n = nBase + t * 16;
    float bv = bias ? bias[n] : 0.0f;
#pragma unroll
    for (int r = 0; r < 8; ++r) {
      int m = mBase + r;
      float val = acc[t][r] + bv;
      if constexpr (EPI == 0) {
        Cf[(long)bl * strideC + (long)m * ldc + n] = val;
      } else if constexpr (EPI == 1) {
        long idx = (long)bl * strideC + (long)m * ldc + n;
        Cf[idx] += val;
      } else if constexpr (EPI == 2) {
        Ch[(long)bl * strideC + (long)m * ldc + n] = (h16)gelu_exact(val);
      } else {  // QKV split
        long rowOff = (long)bl * 512 + m;
        if (n < 256) {
          q16[rowOff * 256 + n] = (h16)val;
        } else if (n < 512) {
          k16[rowOff * 256 + (n - 256)] = (h16)val;
        } else {
          int hh = (n - 512) >> 6, d = (n - 512) & 63;
          vT[(((long)bl * 4 + hh) * 64 + d) * 512 + m] = (h16)val;
        }
      }
    }
  }
}

// ---------------------------------------------------------------------------
// Flash-style attention. grid = (4 m-blocks, 4 heads, 64 seqs), 256 threads.
// Wave w owns 16 query rows; online softmax over 64-token chunks (16 WMMA per
// chunk vs one 16-lane butterfly pair -> tensor-pipe dominated).
// Q,K row-major f16 [seq][tok][h*64+d]; V transposed [seq*4+h][d][tok].
// ---------------------------------------------------------------------------
__global__ __launch_bounds__(256)
void attention_kernel(const h16* __restrict__ q16, const h16* __restrict__ k16,
                      const h16* __restrict__ vT, h16* __restrict__ o16) {
  __shared__ h16 sP[8][16 * 72];  // per-wave 16x64 P tile, padded stride 72
  const int tid = threadIdx.x, w = tid >> 5, lane = tid & 31;
  const int hf = lane >> 4, lm = lane & 15;
  const int bl = blockIdx.z, hh = blockIdx.y;
  const int mBase = blockIdx.x * 128 + w * 16;

  // Q fragments for the two K-steps of dim_head=64 (loaded once).
  F16x16 qf[2];
  const h16* qrow = q16 + ((long)bl * 512 + mBase + lm) * 256 + hh * 64;
#pragma unroll
  for (int s = 0; s < 2; ++s) {
    const h16* p = qrow + s * 32 + hf * 8;
    qf[s].q[0] = *(const uint4*)p;
    qf[s].q[1] = *(const uint4*)(p + 16);
  }

  v8f zero = ZERO8;
  v8f Oacc[4] = {zero, zero, zero, zero};
  float rmax[8], rsum[8];
#pragma unroll
  for (int r = 0; r < 8; ++r) { rmax[r] = -3.0e38f; rsum[r] = 0.f; }

  const h16* kbase = k16 + (long)bl * 512 * 256 + hh * 64;
  const h16* vbase = vT + ((long)bl * 4 + hh) * 64 * 512;

  for (int jc = 0; jc < 512; jc += 64) {
    // Scores S(16 x 64) = Q(16x64) @ K^T: four 16-col tiles, two K-steps each.
    v8f S[4] = {zero, zero, zero, zero};
#pragma unroll
    for (int t = 0; t < 4; ++t) {
#pragma unroll
      for (int s = 0; s < 2; ++s) {
        const h16* krow =
            kbase + (long)(jc + t * 16 + lm) * 256 + s * 32 + hf * 8;
        F16x16 kf;
        kf.q[0] = *(const uint4*)krow;
        kf.q[1] = *(const uint4*)(krow + 16);
        S[t] = wmma_f16(qf[s].v, kf.v, S[t]);
      }
    }
    // Online softmax. Row m = r + 8*hf lives in one 16-lane half-wave; the
    // permlane16 butterfly stays inside each half.
#pragma unroll
    for (int r = 0; r < 8; ++r) {
      float s0 = S[0][r] * 0.125f;  // scale = dim_head^-0.5
      float s1 = S[1][r] * 0.125f;
      float s2 = S[2][r] * 0.125f;
      float s3 = S[3][r] * 0.125f;
      float m = red_max16(fmaxf(fmaxf(s0, s1), fmaxf(s2, s3)));
      float nm = fmaxf(rmax[r], m);
      float corr = __expf(rmax[r] - nm);
      float p0 = __expf(s0 - nm), p1 = __expf(s1 - nm);
      float p2 = __expf(s2 - nm), p3 = __expf(s3 - nm);
      float ps = red_sum16((p0 + p1) + (p2 + p3));
      rsum[r] = rsum[r] * corr + ps;
      rmax[r] = nm;
      // P -> LDS in A-fragment-friendly (row-major) layout.
      h16* pr = &sP[w][(r + 8 * hf) * 72 + lm];
      pr[0]  = (h16)p0;
      pr[16] = (h16)p1;
      pr[32] = (h16)p2;
      pr[48] = (h16)p3;
#pragma unroll
      for (int nt = 0; nt < 4; ++nt) Oacc[nt][r] *= corr;
    }
    // O(16x64) += P(16x64) @ V(64x64): two K-steps x four N-tiles.
#pragma unroll
    for (int ks = 0; ks < 2; ++ks) {
      F16x16 pa;
      const h16* pp = &sP[w][lm * 72 + ks * 32 + hf * 8];
      pa.q[0] = *(const uint4*)pp;
      pa.q[1] = *(const uint4*)(pp + 16);
#pragma unroll
      for (int nt = 0; nt < 4; ++nt) {
        const h16* vrow =
            vbase + (long)(nt * 16 + lm) * 512 + jc + ks * 32 + hf * 8;
        F16x16 vf;
        vf.q[0] = *(const uint4*)vrow;
        vf.q[1] = *(const uint4*)(vrow + 16);
        Oacc[nt] = wmma_f16(pa.v, vf.v, Oacc[nt]);
      }
    }
  }
  // Normalize and emit f16 row-major [seq][tok][h*64+d].
#pragma unroll
  for (int r = 0; r < 8; ++r) {
    int m = mBase + r + 8 * hf;
    float rinv = 1.0f / rsum[r];
    h16* orow = o16 + ((long)bl * 512 + m) * 256 + hh * 64 + lm;
#pragma unroll
    for (int nt = 0; nt < 4; ++nt)
      orow[nt * 16] = (h16)(Oacc[nt][r] * rinv);
  }
}

// ---------------------------------------------------------------------------
// conf[row] = t16[row,:256] . dec_w2 + dec_b2.  One wave per row.
// ---------------------------------------------------------------------------
__global__ __launch_bounds__(256)
void conf_kernel(const h16* __restrict__ t16, const float* __restrict__ w2,
                 const float* __restrict__ b2, float* __restrict__ conf) {
  int w = threadIdx.x >> 5, lane = threadIdx.x & 31;
  long row = (long)blockIdx.x * 8 + w;
  const h16* t = t16 + row * 256;
  float s = 0.f;
#pragma unroll
  for (int i = 0; i < 8; ++i) {
    int c = lane + 32 * i;
    s += (float)t[c] * w2[c];
  }
  s = red_sum32(s);
  if (lane == 0) conf[row] = s + b2[0];
}

// ---------------------------------------------------------------------------
// Per-sequence argmin (first-min tie-break), clamp, gather-compact points,
// and write conf to the second output region.
// ---------------------------------------------------------------------------
__global__ __launch_bounds__(256)
void argmin_gather_kernel(const float* __restrict__ conf,
                          const float* __restrict__ pts,
                          float* __restrict__ out) {
  __shared__ float sv[256];
  __shared__ int si[256];
  __shared__ int minIdx;
  int bl = blockIdx.x, t = threadIdx.x;
  float best = conf[(long)bl * 512 + t];
  int bi = t;
  float c2 = conf[(long)bl * 512 + t + 256];
  if (c2 < best) { best = c2; bi = t + 256; }
  sv[t] = best;
  si[t] = bi;
  __syncthreads();
  for (int s = 128; s > 0; s >>= 1) {
    if (t < s) {
      if (sv[t + s] < sv[t] || (sv[t + s] == sv[t] && si[t + s] < si[t])) {
        sv[t] = sv[t + s];
        si[t] = si[t + s];
      }
    }
    __syncthreads();
  }
  if (t == 0) minIdx = min(si[0], 512 - 64);
  __syncthreads();
  int mi = minIdx;
  // pts output: 448 rows x 3 per sequence, at out[0 .. 86016)
  for (int i = t; i < 448 * 3; i += 256) {
    int r = i / 3, c = i - r * 3;
    int srow = (r < mi) ? r : r + 64;
    out[(long)bl * (448 * 3) + i] = pts[((long)bl * 512 + srow) * 3 + c];
  }
  // conf output at out[86016 .. 118784)
  for (int j = t; j < 512; j += 256)
    out[86016 + (long)bl * 512 + j] = conf[(long)bl * 512 + j];
}

// ---------------------------------------------------------------------------
// Host orchestration.
// ---------------------------------------------------------------------------
extern "C" void kernel_launch(void* const* d_in, const int* in_sizes, int n_in,
                              void* d_out, int out_size, void* d_ws,
                              size_t ws_size, hipStream_t stream) {
  (void)in_sizes; (void)n_in; (void)out_size; (void)ws_size;

  const float* pts      = (const float*)d_in[0];
  const float* enc_w1   = (const float*)d_in[1];
  const float* enc_b1   = (const float*)d_in[2];
  const float* enc_ln_g = (const float*)d_in[3];
  const float* enc_ln_b = (const float*)d_in[4];
  const float* enc_w2   = (const float*)d_in[5];
  const float* enc_b2   = (const float*)d_in[6];
  const float* dec_ln_g = (const float*)d_in[51];
  const float* dec_ln_b = (const float*)d_in[52];
  const float* dec_w1   = (const float*)d_in[53];
  const float* dec_b1   = (const float*)d_in[54];
  const float* dec_w2   = (const float*)d_in[55];
  const float* dec_b2   = (const float*)d_in[56];
  float* out = (float*)d_out;

  char* ws = (char*)d_ws;
  size_t off = 0;
  auto alloc = [&](size_t bytes) -> void* {
    void* p = (void*)(ws + off);
    off += (bytes + 255) & ~(size_t)255;
    return p;
  };

  // f16 transposed weights
  h16* wt_enc2 = (h16*)alloc((size_t)256 * 256 * 2);
  h16* wt_qkv[4]; h16* wt_out[4]; h16* wt_ff1[4]; h16* wt_ff2[4];
  for (int l = 0; l < 4; ++l) {
    wt_qkv[l] = (h16*)alloc((size_t)768 * 256 * 2);
    wt_out[l] = (h16*)alloc((size_t)256 * 256 * 2);
    wt_ff1[l] = (h16*)alloc((size_t)512 * 256 * 2);
    wt_ff2[l] = (h16*)alloc((size_t)256 * 512 * 2);
  }
  h16* wt_dec1 = (h16*)alloc((size_t)256 * 256 * 2);
  // activations
  const long RN = 64L * 512;  // total rows
  float* hbuf = (float*)alloc((size_t)RN * 256 * 4);
  float* conf = (float*)alloc((size_t)RN * 4);
  h16* y16 = (h16*)alloc((size_t)RN * 256 * 2);
  h16* q16 = (h16*)alloc((size_t)RN * 256 * 2);
  h16* k16 = (h16*)alloc((size_t)RN * 256 * 2);
  h16* vTb = (h16*)alloc((size_t)RN * 256 * 2);
  h16* o16 = (h16*)alloc((size_t)RN * 256 * 2);
  h16* t16 = (h16*)alloc((size_t)RN * 512 * 2);

  dim3 blk(256);
  auto eg = [](int n) { return dim3((n + 255) / 256); };

  // Weight prep (transpose + f16)
  transpose_to_f16<<<eg(256 * 256), blk, 0, stream>>>(enc_w2, wt_enc2, 256, 256);
  for (int l = 0; l < 4; ++l) {
    int base = 7 + 11 * l;
    transpose_to_f16<<<eg(256 * 768), blk, 0, stream>>>(
        (const float*)d_in[base + 2], wt_qkv[l], 256, 768);
    transpose_to_f16<<<eg(256 * 256), blk, 0, stream>>>(
        (const float*)d_in[base + 3], wt_out[l], 256, 256);
    transpose_to_f16<<<eg(256 * 512), blk, 0, stream>>>(
        (const float*)d_in[base + 7], wt_ff1[l], 256, 512);
    transpose_to_f16<<<eg(512 * 256), blk, 0, stream>>>(
        (const float*)d_in[base + 9], wt_ff2[l], 512, 256);
  }
  transpose_to_f16<<<eg(256 * 256), blk, 0, stream>>>(dec_w1, wt_dec1, 256, 256);

  const dim3 rowsGrid(4096);                  // 32768 rows / 8 per block
  const long sA256 = 512L * 256, sA512 = 512L * 512;

  // Encoder
  enc_embed_kernel<<<rowsGrid, blk, 0, stream>>>(pts, enc_w1, enc_b1, enc_ln_g,
                                                 enc_ln_b, y16);
  gemm_wmma_kernel<0><<<dim3(4, 4, 64), blk, 0, stream>>>(
      y16, 256, sA256, wt_enc2, 256, enc_b2, hbuf, nullptr, 256, sA256,
      nullptr, nullptr, nullptr);

  // Transformer layers
  for (int l = 0; l < 4; ++l) {
    int base = 7 + 11 * l;
    const float* ln1_g = (const float*)d_in[base + 0];
    const float* ln1_b = (const float*)d_in[base + 1];
    const float* b_out = (const float*)d_in[base + 4];
    const float* ln2_g = (const float*)d_in[base + 5];
    const float* ln2_b = (const float*)d_in[base + 6];
    const float* b_ff1 = (const float*)d_in[base + 8];
    const float* b_ff2 = (const float*)d_in[base + 10];

    ln_kernel<<<rowsGrid, blk, 0, stream>>>(hbuf, ln1_g, ln1_b, y16);
    gemm_wmma_kernel<3><<<dim3(4, 12, 64), blk, 0, stream>>>(
        y16, 256, sA256, wt_qkv[l], 256, nullptr, nullptr, nullptr, 0, 0,
        q16, k16, vTb);
    attention_kernel<<<dim3(4, 4, 64), blk, 0, stream>>>(q16, k16, vTb, o16);
    gemm_wmma_kernel<1><<<dim3(4, 4, 64), blk, 0, stream>>>(
        o16, 256, sA256, wt_out[l], 256, b_out, hbuf, nullptr, 256, sA256,
        nullptr, nullptr, nullptr);
    ln_kernel<<<rowsGrid, blk, 0, stream>>>(hbuf, ln2_g, ln2_b, y16);
    gemm_wmma_kernel<2><<<dim3(4, 8, 64), blk, 0, stream>>>(
        y16, 256, sA256, wt_ff1[l], 256, b_ff1, nullptr, t16, 512, sA512,
        nullptr, nullptr, nullptr);
    gemm_wmma_kernel<1><<<dim3(4, 4, 64), blk, 0, stream>>>(
        t16, 512, sA512, wt_ff2[l], 512, b_ff2, hbuf, nullptr, 256, sA256,
        nullptr, nullptr, nullptr);
  }

  // Decoder
  ln_kernel<<<rowsGrid, blk, 0, stream>>>(hbuf, dec_ln_g, dec_ln_b, y16);
  gemm_wmma_kernel<2><<<dim3(4, 4, 64), blk, 0, stream>>>(
      y16, 256, sA256, wt_dec1, 256, dec_b1, nullptr, t16, 256, sA256,
      nullptr, nullptr, nullptr);
  conf_kernel<<<rowsGrid, blk, 0, stream>>>(t16, dec_w2, dec_b2, conf);

  // Removal: argmin + gather compaction, plus conf output.
  argmin_gather_kernel<<<dim3(64), blk, 0, stream>>>(conf, pts, out);
}
